// AttentionLayer_3874060501769
// MI455X (gfx1250) — compile-verified, matrix operand order checked by reading, not run on hardware
//
#include <hip/hip_runtime.h>
#include <hip/hip_bf16.h>
#include <math.h>

typedef __attribute__((ext_vector_type(16))) _Float16 v16h;
typedef __attribute__((ext_vector_type(8)))  _Float16 v8h;
typedef __attribute__((ext_vector_type(8)))  float    v8f;

union HF { v16h v; v8h h[2]; };

static constexpr int kDim  = 1024;
static constexpr int kHeads = 16;
static constexpr int kHd   = 64;
static constexpr int kB    = 2;
static constexpr int kT    = 2048;

// CDNA5 async copy: 16 bytes/lane, global -> LDS, tracked by ASYNCcnt.
#define ASYNC_G2L_B128(ldsOff, gPtr)                                     \
  asm volatile("global_load_async_to_lds_b128 %0, %1, off"               \
               :: "v"(ldsOff), "v"(gPtr) : "memory")
#define WAIT_ASYNC0() asm volatile("s_wait_asynccnt 0x0" ::: "memory")

__device__ __forceinline__ unsigned lds_off(const void* p) {
  // generic shared pointer: addr[31:0] == LDS offset (ISA 10.2 aperture rules)
  return (unsigned)(unsigned long long)p;
}

// ---------------------------------------------------------------------------
// fp32 -> fp16 convert
// ---------------------------------------------------------------------------
__global__ void f32_to_f16_kernel(const float* __restrict__ in,
                                  _Float16* __restrict__ out, int n) {
  int i = blockIdx.x * blockDim.x + threadIdx.x;
  if (i < n) out[i] = (_Float16)in[i];
}

// ---------------------------------------------------------------------------
// Tiled WMMA GEMM:  C[M,N] = A[M,K] * W[N,K]^T + bias
// Block tile 64(M) x 128(N) x 64(K), 256 threads = 8 waves (2 x 4),
// each wave 32x32.  A/B staged with GLOBAL_LOAD_ASYNC_TO_LDS_B128.
// ---------------------------------------------------------------------------
__global__ __launch_bounds__(256)
void gemm_wmma_kernel(const _Float16* __restrict__ A,
                      const _Float16* __restrict__ W,
                      const float* __restrict__ bias,
                      _Float16* __restrict__ outH,
                      float* __restrict__ outF,
                      int M, int N, int K) {
  __shared__ _Float16 As[64 * 72];    // 64 cols + 8 pad
  __shared__ _Float16 Bs[128 * 72];

  const int tid  = threadIdx.x;
  const int lane = tid & 31;
  const int w    = tid >> 5;
  const int ln   = lane & 15;
  const int hk   = lane >> 4;
  const int mi   = w & 1;        // 2 m-subtiles of 32
  const int ni   = w >> 1;       // 4 n-subtiles of 32
  const int mB   = blockIdx.y * 64;
  const int nB   = blockIdx.x * 128;

  const v8f zf = {0.f, 0.f, 0.f, 0.f, 0.f, 0.f, 0.f, 0.f};
  v8f c[2][2];
  c[0][0] = zf; c[0][1] = zf; c[1][0] = zf; c[1][1] = zf;

  const int r8 = tid >> 3;            // 0..31
  const int cg = (tid & 7) * 8;       // 0,8,...,56  (16B chunks over 64 cols)

  const unsigned lA0 = lds_off(&As[(r8)      * 72 + cg]);
  const unsigned lA1 = lds_off(&As[(r8 + 32) * 72 + cg]);
  const unsigned lB0 = lds_off(&Bs[(r8)      * 72 + cg]);
  const unsigned lB1 = lds_off(&Bs[(r8 + 32) * 72 + cg]);
  const unsigned lB2 = lds_off(&Bs[(r8 + 64) * 72 + cg]);
  const unsigned lB3 = lds_off(&Bs[(r8 + 96) * 72 + cg]);

  for (int k0 = 0; k0 < K; k0 += 64) {
    // burst of 6 async copies per thread: 64x64 A tile + 128x64 B tile
    ASYNC_G2L_B128(lA0, A + (size_t)(mB + r8)      * K + k0 + cg);
    ASYNC_G2L_B128(lA1, A + (size_t)(mB + r8 + 32) * K + k0 + cg);
    ASYNC_G2L_B128(lB0, W + (size_t)(nB + r8)      * K + k0 + cg);
    ASYNC_G2L_B128(lB1, W + (size_t)(nB + r8 + 32) * K + k0 + cg);
    ASYNC_G2L_B128(lB2, W + (size_t)(nB + r8 + 64) * K + k0 + cg);
    ASYNC_G2L_B128(lB3, W + (size_t)(nB + r8 + 96) * K + k0 + cg);
    if (k0 + 64 < K) {  // prefetch next K-slab into L2
      __builtin_prefetch(A + (size_t)(mB + r8) * K + k0 + 64 + cg, 0, 1);
      __builtin_prefetch(W + (size_t)(nB + r8) * K + k0 + 64 + cg, 0, 1);
    }
    WAIT_ASYNC0();
    __syncthreads();

#pragma unroll
    for (int ks = 0; ks < 64; ks += 32) {
      HF a[2], b[2];
#pragma unroll
      for (int f = 0; f < 2; ++f) {
        const int m = mi * 32 + f * 16 + ln;
        a[f].h[0] = *(const v8h*)(&As[m * 72 + ks + hk * 8]);
        a[f].h[1] = *(const v8h*)(&As[m * 72 + ks + 16 + hk * 8]);
        const int n = ni * 32 + f * 16 + ln;
        b[f].h[0] = *(const v8h*)(&Bs[n * 72 + ks + hk * 8]);
        b[f].h[1] = *(const v8h*)(&Bs[n * 72 + ks + 16 + hk * 8]);
      }
#pragma unroll
      for (int x = 0; x < 2; ++x)
#pragma unroll
        for (int y = 0; y < 2; ++y)
          c[x][y] = __builtin_amdgcn_wmma_f32_16x16x32_f16(
              false, a[x].v, false, b[y].v, (short)0, c[x][y], false, false);
    }
    __syncthreads();
  }

#pragma unroll
  for (int x = 0; x < 2; ++x) {
#pragma unroll
    for (int y = 0; y < 2; ++y) {
      const int n  = nB + ni * 32 + y * 16 + ln;
      const float bv = bias ? bias[n] : 0.f;
#pragma unroll
      for (int r = 0; r < 8; ++r) {
        const int m = mB + mi * 32 + x * 16 + r + 8 * hk;
        const float v = c[x][y][r] + bv;
        if (outF) outF[(size_t)m * N + n] = v;
        else      outH[(size_t)m * N + n] = (_Float16)v;
      }
    }
  }
}

// ---------------------------------------------------------------------------
// RoPE + head split.  QKV[B*T, 3*DIM] f16 -> Q,K [B,H,T,Dh] f16 (rotated),
// V transposed to [B,H,Dh,T] f16 so PV B-fragments are contiguous.
// ---------------------------------------------------------------------------
__global__ void rope_split_kernel(const _Float16* __restrict__ QKV,
                                  _Float16* __restrict__ Q,
                                  _Float16* __restrict__ Kk,
                                  _Float16* __restrict__ Vt) {
  const int idx = blockIdx.x * blockDim.x + threadIdx.x;
  if (idx >= kB * kT * kHeads * 32) return;
  const int j = idx & 31;
  const int h = (idx >> 5) & 15;
  const int t = (idx >> 9) & (kT - 1);
  const int b = idx >> 20;

  const _Float16* row = QKV + (size_t)(b * kT + t) * (3 * kDim);
  const int e = 2 * j, o = 2 * j + 1;
  const float qe = (float)row[h * kHd + e],            qo = (float)row[h * kHd + o];
  const float ke = (float)row[kDim + h * kHd + e],     ko = (float)row[kDim + h * kHd + o];
  const float ve = (float)row[2 * kDim + h * kHd + e], vo = (float)row[2 * kDim + h * kHd + o];

  const int fi = (2 * j) & 31;                       // inv_freq index of even slot
  const float inv = __expf(-(float)fi * 0.28782313662425574f);  // ln(1e4)/32
  float s, c;
  __sincosf((float)t * inv, &s, &c);

  const size_t qoff = ((size_t)(b * kHeads + h) * kT + t) * kHd;
  Q[qoff + e]  = (_Float16)(qe * c - qo * s);
  Q[qoff + o]  = (_Float16)(qe * s + qo * c);
  Kk[qoff + e] = (_Float16)(ke * c - ko * s);
  Kk[qoff + o] = (_Float16)(ke * s + ko * c);

  const size_t vtb = (size_t)(b * kHeads + h) * kHd;
  Vt[(vtb + e) * kT + t] = (_Float16)ve;
  Vt[(vtb + o) * kT + t] = (_Float16)vo;
}

// ---------------------------------------------------------------------------
// Flash attention, WMMA f16 -> f32.  Grid (T/64, H, B), 4 waves/WG, each wave
// owns 16 query rows.  Online softmax over 32-key blocks; P goes through a
// per-wave LDS tile to convert C-layout -> A-layout for the PV WMMA.
// ---------------------------------------------------------------------------
__global__ __launch_bounds__(128)
void flash_attn_wmma_kernel(const _Float16* __restrict__ Q,
                            const _Float16* __restrict__ Kk,
                            const _Float16* __restrict__ Vt,
                            const unsigned char* __restrict__ amask,
                            _Float16* __restrict__ O) {
  __shared__ _Float16 Ps[4][16 * 40];   // per-wave 16x32 P tile (+8 pad)

  const int lane = threadIdx.x & 31;
  const int w    = threadIdx.x >> 5;
  const int ln   = lane & 15;
  const int hk   = lane >> 4;
  const int q0   = blockIdx.x * 64 + w * 16;
  const int h    = blockIdx.y;
  const int b    = blockIdx.z;

  const size_t bh = (size_t)(b * kHeads + h);
  const _Float16* qb = Q  + bh * kT * kHd;
  const _Float16* kp = Kk + bh * kT * kHd;
  const _Float16* vb = Vt + bh * kHd * kT;
  const unsigned char* mk = amask + (size_t)b * kT;

  // Q fragments for this wave's 16 rows, kept in registers (K = 0..63).
  HF qf[2];
  {
    const _Float16* p = qb + (size_t)(q0 + ln) * kHd;
    const int base = hk * 8;
    qf[0].h[0] = *(const v8h*)(p + base);
    qf[0].h[1] = *(const v8h*)(p + 16 + base);
    qf[1].h[0] = *(const v8h*)(p + 32 + base);
    qf[1].h[1] = *(const v8h*)(p + 48 + base);
  }

  const v8f zf = {0.f, 0.f, 0.f, 0.f, 0.f, 0.f, 0.f, 0.f};
  v8f o[4];  o[0] = zf; o[1] = zf; o[2] = zf; o[3] = zf;
  float mi[8], li[8];
#pragma unroll
  for (int r = 0; r < 8; ++r) { mi[r] = -__builtin_inff(); li[r] = 0.f; }

  _Float16* pt = Ps[w];
  const int nkb = (q0 + 15) / 32 + 1;   // causal: last 32-key block touched

  for (int kb = 0; kb < nkb; ++kb) {
    const int k0 = kb * 32;
#pragma unroll
    for (int nh = 0; nh < 2; ++nh) {
      // --- S = Q * K^T for a 16x16 tile (two chained k32 WMMAs) ---
      const int key = k0 + nh * 16 + ln;
      const _Float16* krow = kp + (size_t)key * kHd;
      const int base = hk * 8;
      HF kf0, kf1;
      kf0.h[0] = *(const v8h*)(krow + base);
      kf0.h[1] = *(const v8h*)(krow + 16 + base);
      kf1.h[0] = *(const v8h*)(krow + 32 + base);
      kf1.h[1] = *(const v8h*)(krow + 48 + base);

      v8f s = zf;
      s = __builtin_amdgcn_wmma_f32_16x16x32_f16(false, qf[0].v, false, kf0.v,
                                                 (short)0, s, false, false);
      s = __builtin_amdgcn_wmma_f32_16x16x32_f16(false, qf[1].v, false, kf1.v,
                                                 (short)0, s, false, false);

      const int ncol  = k0 + nh * 16 + ln;
      const bool kok  = mk[ncol] != 0;
#pragma unroll
      for (int r = 0; r < 8; ++r) {
        const int mrow = q0 + r + 8 * hk;
        float sv = s[r] * 0.125f;                       // Dh^-0.5
        if (ncol > mrow || !kok) sv = -__builtin_inff();
        // row max over the 16 lanes of this half-wave
        float bm = sv;
        bm = fmaxf(bm, __shfl_xor(bm, 1, 32));
        bm = fmaxf(bm, __shfl_xor(bm, 2, 32));
        bm = fmaxf(bm, __shfl_xor(bm, 4, 32));
        bm = fmaxf(bm, __shfl_xor(bm, 8, 32));
        const float nm  = fmaxf(mi[r], bm);
        const float fac = __expf(mi[r] - nm);
        const float pv  = __expf(sv - nm);
        float bs = pv;
        bs += __shfl_xor(bs, 1, 32);
        bs += __shfl_xor(bs, 2, 32);
        bs += __shfl_xor(bs, 4, 32);
        bs += __shfl_xor(bs, 8, 32);
        li[r] = li[r] * fac + bs;
        mi[r] = nm;
#pragma unroll
        for (int f = 0; f < 4; ++f) o[f][r] *= fac;
        pt[(r + 8 * hk) * 40 + nh * 16 + ln] = (_Float16)pv;   // C -> LDS
      }
    }
    // make P tile visible to all lanes of this wave before re-reading
    asm volatile("s_wait_dscnt 0x0" ::: "memory");

    // P as A-fragment (16x32)
    HF pf;
    pf.h[0] = *(const v8h*)(pt + ln * 40 + hk * 8);
    pf.h[1] = *(const v8h*)(pt + ln * 40 + 16 + hk * 8);

    // O += P * V  (V pre-transposed: column d of V is contiguous over keys)
#pragma unroll
    for (int f = 0; f < 4; ++f) {
      const int d = f * 16 + ln;
      const _Float16* vp = vb + (size_t)d * kT + k0;
      HF vf;
      vf.h[0] = *(const v8h*)(vp + hk * 8);
      vf.h[1] = *(const v8h*)(vp + 16 + hk * 8);
      o[f] = __builtin_amdgcn_wmma_f32_16x16x32_f16(false, pf.v, false, vf.v,
                                                    (short)0, o[f], false, false);
    }
    asm volatile("s_wait_dscnt 0x0" ::: "memory");  // WAR guard before next stores
  }

  // epilogue: divide by row sums, write [B,T,H*Dh] f16
#pragma unroll
  for (int f = 0; f < 4; ++f) {
    const int d = f * 16 + ln;
#pragma unroll
    for (int r = 0; r < 8; ++r) {
      const int t = q0 + r + 8 * hk;
      const float val = o[f][r] / li[r];
      O[((size_t)(b * kT + t)) * kDim + h * kHd + d] = (_Float16)val;
    }
  }
}

// ---------------------------------------------------------------------------
// launch
// ---------------------------------------------------------------------------
extern "C" void kernel_launch(void* const* d_in, const int* in_sizes, int n_in,
                              void* d_out, int out_size, void* d_ws, size_t ws_size,
                              hipStream_t stream) {
  const float*         x      = (const float*)d_in[0];
  const unsigned char* amask  = (const unsigned char*)d_in[1];
  const float*         qkv_w  = (const float*)d_in[2];
  const float*         qkv_b  = (const float*)d_in[3];
  const float*         out_w  = (const float*)d_in[4];
  const float*         out_b  = (const float*)d_in[5];
  float*               out    = (float*)d_out;

  char* ws = (char*)d_ws;
  const size_t nX   = (size_t)kB * kT * kDim;        // 4,194,304
  const size_t nWq  = (size_t)3 * kDim * kDim;       // 3,145,728
  const size_t nWo  = (size_t)kDim * kDim;           // 1,048,576
  const size_t nQKV = (size_t)kB * kT * 3 * kDim;    // 12,582,912
  const size_t nQ   = (size_t)kB * kHeads * kT * kHd;// 4,194,304

  _Float16* Xh   = (_Float16*)(ws);
  _Float16* Wqh  = (_Float16*)(ws + nX * 2);
  _Float16* Woh  = (_Float16*)(ws + nX * 2 + nWq * 2);
  _Float16* QKVh = (_Float16*)(ws + nX * 2 + nWq * 2 + nWo * 2);
  _Float16* Qh   = (_Float16*)((char*)QKVh + nQKV * 2);
  _Float16* Kh   = (_Float16*)((char*)Qh + nQ * 2);
  _Float16* Vth  = (_Float16*)((char*)Kh + nQ * 2);
  _Float16* Oh   = (_Float16*)((char*)Vth + nQ * 2);

  f32_to_f16_kernel<<<(int)((nX  + 255) / 256), 256, 0, stream>>>(x, Xh, (int)nX);
  f32_to_f16_kernel<<<(int)((nWq + 255) / 256), 256, 0, stream>>>(qkv_w, Wqh, (int)nWq);
  f32_to_f16_kernel<<<(int)((nWo + 255) / 256), 256, 0, stream>>>(out_w, Woh, (int)nWo);

  // QKV projection: [4096,3072] = Xh[4096,1024] * Wq[3072,1024]^T + b
  gemm_wmma_kernel<<<dim3(3 * kDim / 128, kB * kT / 64), 256, 0, stream>>>(
      Xh, Wqh, qkv_b, QKVh, nullptr, kB * kT, 3 * kDim, kDim);

  rope_split_kernel<<<(kB * kT * kHeads * 32) / 256, 256, 0, stream>>>(
      QKVh, Qh, Kh, Vth);

  flash_attn_wmma_kernel<<<dim3(kT / 64, kHeads, kB), 128, 0, stream>>>(
      Qh, Kh, Vth, amask, Oh);

  // output projection: [4096,1024] = Oh[4096,1024] * Wo[1024,1024]^T + b
  gemm_wmma_kernel<<<dim3(kDim / 128, kB * kT / 64), 256, 0, stream>>>(
      Oh, Woh, out_b, nullptr, out, kB * kT, kDim, kDim);
}
